// Attention_4294967296060
// MI455X (gfx1250) — matrix operand order repaired and checked by reading, not compiled, not run on hardware
//
#include <hip/hip_runtime.h>
#include <math.h>

#define D      128
#define TPB    256     // 8 wave32
#define TILE   4       // 16-row strips per wave
#define ROWS   (8 * 16 * TILE)   // 512 rows per block
#define PADW   136     // padded LDS row length (ushorts)

typedef __bf16 bf16_t;
typedef bf16_t v16bf __attribute__((ext_vector_type(16)));
typedef float  v8f   __attribute__((ext_vector_type(8)));

union BF8x { unsigned int u[8]; v16bf v; };

__device__ __forceinline__ unsigned short f32_to_bf16_rne(float a) {
  unsigned int u = __builtin_bit_cast(unsigned int, a);
  u += 0x7FFFu + ((u >> 16) & 1u);
  return (unsigned short)(u >> 16);
}
__device__ __forceinline__ float bf16_hi_to_f32(unsigned short h) {
  unsigned int u = ((unsigned int)h) << 16;
  return __builtin_bit_cast(float, u);
}
// a ~= hi + lo (bf16 each): bf16x3 fp32 emulation on WMMA
__device__ __forceinline__ void split2(float a, unsigned short& hi, unsigned short& lo) {
  unsigned short h = f32_to_bf16_rne(a);
  lo = f32_to_bf16_rne(a - bf16_hi_to_f32(h));
  hi = h;
}

// per-ISA per-lane K index inside a 32-wide chunk
__device__ __forceinline__ int kA_idx(int i, int hf) {  // A 16x32 16-bit layout
  return ((i < 4) ? 2 * i : 16 + 2 * (i - 4)) + 8 * hf;
}
__device__ __forceinline__ int kB_idx(int i, int hf) {  // B 32x16 16-bit layout
  return ((i < 4) ? 2 * i : 16 + 2 * (i - 4)) + 8 * hf;
}

// ---------------- prep: split weights to bf16 hi/lo once; compute c1 once ----
__global__ __launch_bounds__(TPB)
void prep_kernel(const float* __restrict__ W1, const float* __restrict__ b1,
                 const float* __restrict__ W2, const float* __restrict__ u_rep,
                 unsigned short* __restrict__ w1hi, unsigned short* __restrict__ w1lo,
                 unsigned short* __restrict__ w2hi, unsigned short* __restrict__ w2lo,
                 float* __restrict__ c1) {
  const int gid = blockIdx.x * TPB + threadIdx.x;
  const int nthr = gridDim.x * TPB;
  for (int idx = gid; idx < D * D; idx += nthr) {
    int f = idx >> 7, k = idx & 127;
    split2(W1[f * 2 * D + k], w1hi[idx], w1lo[idx]);   // left half of W1
    split2(W2[f * D + k],     w2hi[idx], w2lo[idx]);
  }
  if (blockIdx.x == 0 && threadIdx.x < D) {
    int f = threadIdx.x;
    float acc = b1[f];
    const float* wr = W1 + f * 2 * D + D;              // right half of W1
    for (int k = 0; k < D; ++k) acc = fmaf(wr[k], u_rep[k], acc);
    c1[f] = acc;
  }
}

// ---------------- main: fused 2-layer MLP + score, WMMA bf16x3 ---------------
__global__ __launch_bounds__(TPB)
void att_mlp_kernel(const float* __restrict__ node1,
                    const unsigned short* __restrict__ w1hi,
                    const unsigned short* __restrict__ w1lo,
                    const unsigned short* __restrict__ w2hi,
                    const unsigned short* __restrict__ w2lo,
                    const float* __restrict__ c1g,
                    const float* __restrict__ b2, const float* __restrict__ W3,
                    const float* __restrict__ b3,
                    float* __restrict__ s_out, float* __restrict__ bmax,
                    int N)
{
  __shared__ unsigned short sW1hi[D][PADW];
  __shared__ unsigned short sW1lo[D][PADW];
  __shared__ unsigned short sW2hi[D][PADW];
  __shared__ unsigned short sW2lo[D][PADW];
  __shared__ unsigned short sH1hi[128][PADW];   // per-wave 16-row regions, reused per strip
  __shared__ unsigned short sH1lo[128][PADW];
  __shared__ float sC1[D], sB2[D], sW3[D], sS[ROWS];

  const int tid = threadIdx.x;

  // ---- async-copy pre-split weights into padded LDS (16B chunks) ----
  // chunk q of matrix m: row = q>>4, 8-ushort group = q&15
  {
    const unsigned short* gsrc[4] = { w1hi, w1lo, w2hi, w2lo };
    unsigned lbase[4] = {
      (unsigned)(size_t)&sW1hi[0][0], (unsigned)(size_t)&sW1lo[0][0],
      (unsigned)(size_t)&sW2hi[0][0], (unsigned)(size_t)&sW2lo[0][0] };
#pragma unroll
    for (int m = 0; m < 4; ++m) {
      for (int q = tid; q < D * 16; q += TPB) {
        int row = q >> 4, g = q & 15;
        unsigned lds_off = lbase[m] + (unsigned)(row * PADW + g * 8) * 2u;
        unsigned long long gaddr =
            (unsigned long long)(size_t)(gsrc[m] + (size_t)(row * D + g * 8));
        asm volatile("global_load_async_to_lds_b128 %0, %1, off"
                     :: "v"(lds_off), "v"(gaddr) : "memory");
      }
    }
  }
  for (int i = tid; i < D; i += TPB) { sC1[i] = c1g[i]; sB2[i] = b2[i]; sW3[i] = W3[i]; }
  asm volatile("s_wait_asynccnt 0x0" ::: "memory");
  __syncthreads();

  const int wave = tid >> 5;
  const int lane = tid & 31;
  const int n    = lane & 15;
  const int hf   = lane >> 4;
  const int myRow = wave * 16;        // wave's sH1 region
  const float b3v = b3[0];

  for (int t = 0; t < TILE; ++t) {
    const int strip   = wave * TILE + t;
    const int rowBase = blockIdx.x * ROWS + strip * 16;

    // ---- load + split this strip's 16x128 A (masked, non-divergent) ----
    const int  rA    = rowBase + n;
    const int  rClmp = (rA < N) ? rA : (N - 1);
    const float mask = (rA < N) ? 1.0f : 0.0f;
    const float* arow = node1 + (size_t)rClmp * D;

    BF8x Ahi[4], Alo[4];
#pragma unroll
    for (int c = 0; c < 4; ++c) {
#pragma unroll
      for (int g = 0; g < 2; ++g) {               // i = 4g..4g+3 : 8 contiguous floats
        const int kb = c * 32 + g * 16 + 8 * hf;
        float4 x0 = *(const float4*)(arow + kb);
        float4 x1 = *(const float4*)(arow + kb + 4);
        float fv[8] = { x0.x, x0.y, x0.z, x0.w, x1.x, x1.y, x1.z, x1.w };
#pragma unroll
        for (int m = 0; m < 4; ++m) {
          unsigned short h0, l0, h1, l1;
          split2(fv[2 * m] * mask, h0, l0);
          split2(fv[2 * m + 1] * mask, h1, l1);
          Ahi[c].u[4 * g + m] = ((unsigned int)h1 << 16) | h0;
          Alo[c].u[4 * g + m] = ((unsigned int)l1 << 16) | l0;
        }
      }
    }

    // ---- layer 1: h1 = relu(A . W1a^T + c1), bf16x3 ----
    for (int j = 0; j < 8; ++j) {
      v8f acc = {0.f, 0.f, 0.f, 0.f, 0.f, 0.f, 0.f, 0.f};
#pragma unroll
      for (int c = 0; c < 4; ++c) {
        BF8x Bhi, Blo;
        const int f = j * 16 + n;
#pragma unroll
        for (int i = 0; i < 8; ++i) {
          int k = c * 32 + kB_idx(i, hf);
          Bhi.u[i] = *(const unsigned int*)&sW1hi[f][k];
          Blo.u[i] = *(const unsigned int*)&sW1lo[f][k];
        }
        acc = __builtin_amdgcn_wmma_f32_16x16x32_bf16(false, Ahi[c].v, false, Bhi.v, (short)0, acc, false, false);
        acc = __builtin_amdgcn_wmma_f32_16x16x32_bf16(false, Ahi[c].v, false, Blo.v, (short)0, acc, false, false);
        acc = __builtin_amdgcn_wmma_f32_16x16x32_bf16(false, Alo[c].v, false, Bhi.v, (short)0, acc, false, false);
      }
      const float cc = sC1[j * 16 + n];
#pragma unroll
      for (int v = 0; v < 8; ++v) {
        float h = acc[v] + cc;
        h = h > 0.f ? h : 0.f;
        unsigned short hh, hl;
        split2(h, hh, hl);
        int rr = myRow + v + 8 * hf;           // C/D layout: M = v + 8*half
        sH1hi[rr][j * 16 + n] = hh;
        sH1lo[rr][j * 16 + n] = hl;
      }
    }
    // same-wave LDS produce->consume: DS ops in-order per wave (dscnt waits)

    // ---- layer 2 A operand from LDS ----
    BF8x A2hi[4], A2lo[4];
#pragma unroll
    for (int c = 0; c < 4; ++c) {
#pragma unroll
      for (int i = 0; i < 8; ++i) {
        int k = c * 32 + kA_idx(i, hf);
        int rr = myRow + n;
        A2hi[c].u[i] = *(const unsigned int*)&sH1hi[rr][k];
        A2lo[c].u[i] = *(const unsigned int*)&sH1lo[rr][k];
      }
    }

    // ---- layer 2 + fused layer-3 dot with w3 ----
    float p[8];
#pragma unroll
    for (int v = 0; v < 8; ++v) p[v] = 0.f;

    for (int j = 0; j < 8; ++j) {
      v8f acc = {0.f, 0.f, 0.f, 0.f, 0.f, 0.f, 0.f, 0.f};
#pragma unroll
      for (int c = 0; c < 4; ++c) {
        BF8x Bhi, Blo;
        const int f = j * 16 + n;
#pragma unroll
        for (int i = 0; i < 8; ++i) {
          int k = c * 32 + kB_idx(i, hf);
          Bhi.u[i] = *(const unsigned int*)&sW2hi[f][k];
          Blo.u[i] = *(const unsigned int*)&sW2lo[f][k];
        }
        acc = __builtin_amdgcn_wmma_f32_16x16x32_bf16(false, A2hi[c].v, false, Bhi.v, (short)0, acc, false, false);
        acc = __builtin_amdgcn_wmma_f32_16x16x32_bf16(false, A2hi[c].v, false, Blo.v, (short)0, acc, false, false);
        acc = __builtin_amdgcn_wmma_f32_16x16x32_bf16(false, A2lo[c].v, false, Bhi.v, (short)0, acc, false, false);
      }
      const float bb  = sB2[j * 16 + n];
      const float w3v = sW3[j * 16 + n];
#pragma unroll
      for (int v = 0; v < 8; ++v) {
        float h = acc[v] + bb;
        h = h > 0.f ? h : 0.f;
        p[v] = fmaf(h, w3v, p[v]);
      }
    }

    // butterfly sum over the 16 lanes sharing a half (all 128 cols per row)
#pragma unroll
    for (int v = 0; v < 8; ++v) {
      p[v] += __shfl_xor(p[v], 1);
      p[v] += __shfl_xor(p[v], 2);
      p[v] += __shfl_xor(p[v], 4);
      p[v] += __shfl_xor(p[v], 8);
    }

    if (n == 0) {          // lanes 0 (rows 0..7) and 16 (rows 8..15)
#pragma unroll
      for (int v = 0; v < 8; ++v) {
        int rloc = v + 8 * hf;
        int grow = rowBase + rloc;
        float sval = p[v] + b3v;
        if (grow < N) { s_out[grow] = sval; sS[strip * 16 + rloc] = sval; }
        else          { sS[strip * 16 + rloc] = -INFINITY; }
      }
    }
  }

  __syncthreads();
  // block max over ROWS=512 scores (deterministic tree)
  sS[tid] = fmaxf(sS[tid], sS[tid + 256]);
  __syncthreads();
  for (int st = 128; st > 0; st >>= 1) {
    if (tid < st) sS[tid] = fmaxf(sS[tid], sS[tid + st]);
    __syncthreads();
  }
  if (tid == 0) bmax[blockIdx.x] = sS[0];
}

// ---------------- softmax pipeline (deterministic, graph-safe) ----------------
__global__ void reduce_max_kernel(const float* __restrict__ bmax, int nb,
                                  float* __restrict__ gmax) {
  __shared__ float sm[256];
  float m = -INFINITY;
  for (int i = threadIdx.x; i < nb; i += 256) m = fmaxf(m, bmax[i]);
  sm[threadIdx.x] = m; __syncthreads();
  for (int s = 128; s > 0; s >>= 1) {
    if (threadIdx.x < s) sm[threadIdx.x] = fmaxf(sm[threadIdx.x], sm[threadIdx.x + s]);
    __syncthreads();
  }
  if (threadIdx.x == 0) gmax[0] = sm[0];
}

__global__ void exp_kernel(float* __restrict__ s, int N,
                           const float* __restrict__ gmax,
                           float* __restrict__ bsum) {
  __shared__ float sm[256];
  int i = blockIdx.x * 256 + threadIdx.x;
  float e = 0.f;
  if (i < N) { e = expf(s[i] - gmax[0]); s[i] = e; }
  sm[threadIdx.x] = e; __syncthreads();
  for (int st = 128; st > 0; st >>= 1) {
    if (threadIdx.x < st) sm[threadIdx.x] += sm[threadIdx.x + st];
    __syncthreads();
  }
  if (threadIdx.x == 0) bsum[blockIdx.x] = sm[0];
}

__global__ void reduce_sum_kernel(const float* __restrict__ bsum, int nb,
                                  float* __restrict__ total) {
  __shared__ float sm[256];
  float t = 0.f;
  for (int i = threadIdx.x; i < nb; i += 256) t += bsum[i];
  sm[threadIdx.x] = t; __syncthreads();
  for (int s = 128; s > 0; s >>= 1) {
    if (threadIdx.x < s) sm[threadIdx.x] += sm[threadIdx.x + s];
    __syncthreads();
  }
  if (threadIdx.x == 0) total[0] = sm[0];
}

__global__ void norm_kernel(float* __restrict__ e, int N,
                            const float* __restrict__ total) {
  int i = blockIdx.x * 256 + threadIdx.x;
  if (i < N) e[i] = e[i] / total[0];
}

extern "C" void kernel_launch(void* const* d_in, const int* in_sizes, int n_in,
                              void* d_out, int out_size, void* d_ws, size_t ws_size,
                              hipStream_t stream) {
  (void)in_sizes; (void)n_in; (void)ws_size;
  const float* node1 = (const float*)d_in[0];
  const float* u_rep = (const float*)d_in[1];
  // d_in[2] = num_neighs scalar (unused; N taken from out_size)
  const float* W1 = (const float*)d_in[3];
  const float* b1 = (const float*)d_in[4];
  const float* W2 = (const float*)d_in[5];
  const float* b2 = (const float*)d_in[6];
  const float* W3 = (const float*)d_in[7];
  const float* b3 = (const float*)d_in[8];
  float* out = (float*)d_out;

  const int N   = out_size;               // scores live in d_out (in-place softmax)
  const int nb  = (N + ROWS - 1) / ROWS;  // main-kernel blocks (512 rows each)
  const int nb2 = (N + 255) / 256;        // elementwise blocks

  // workspace layout: 4 bf16 matrices (32KB each), then floats
  unsigned short* wsu = (unsigned short*)d_ws;
  unsigned short* w1hi = wsu;
  unsigned short* w1lo = wsu + D * D;
  unsigned short* w2hi = wsu + 2 * D * D;
  unsigned short* w2lo = wsu + 3 * D * D;
  float* wsf   = (float*)(wsu + 4 * D * D);
  float* c1    = wsf;                     // [128]
  float* gmax  = wsf + D;                 // [1]
  float* total = wsf + D + 1;             // [1]
  float* bmaxp = wsf + D + 2;             // [nb]
  float* bsum  = wsf + D + 2 + nb;        // [nb2]

  prep_kernel<<<16, TPB, 0, stream>>>(W1, b1, W2, u_rep, w1hi, w1lo, w2hi, w2lo, c1);
  att_mlp_kernel<<<nb, TPB, 0, stream>>>(node1, w1hi, w1lo, w2hi, w2lo, c1,
                                         b2, W3, b3, out, bmaxp, N);
  reduce_max_kernel<<<1, 256, 0, stream>>>(bmaxp, nb, gmax);
  exp_kernel<<<nb2, 256, 0, stream>>>(out, N, gmax, bsum);
  reduce_sum_kernel<<<1, 256, 0, stream>>>(bsum, nb2, total);
  norm_kernel<<<nb2, 256, 0, stream>>>(out, N, total);
}
